// MultiHeadInfiniAttention_6098853560554
// MI455X (gfx1250) — matrix operand order repaired and checked by reading, not compiled, not run on hardware
//
#include <hip/hip_runtime.h>
#include <cstdint>

// ---------------- problem constants ----------------
constexpr int NH     = 8;      // heads
constexpr int DK     = 128;    // key dim per head
constexpr int DV     = 128;    // value dim per head
constexpr int SEGL   = 512;    // segment length
constexpr int NSEG   = 16;     // 8192 / 512
constexpr int BB     = 4;      // batch
constexpr int SS     = 8192;   // sequence
constexpr int DMODEL = 1024;   // input dim == NH*DK
constexpr int MTOT   = BB * SS;

typedef __bf16 bf16_t;
typedef __attribute__((ext_vector_type(16))) __bf16 v16bf;
typedef __attribute__((ext_vector_type(8)))  float  v8f;

__device__ __forceinline__ int lane_id() { return threadIdx.x & 31; }

__device__ __forceinline__ v8f wmma_bf16(v16bf a, v16bf b, v8f c) {
  // D = A(16x32 bf16) * B(32x16 bf16) + C(16x16 f32)
  return __builtin_amdgcn_wmma_f32_16x16x32_bf16(false, a, false, b, (short)0, c,
                                                 false, false);
}

__device__ __forceinline__ float elup1(float x) {  // elu(x)+1
  return x > 0.f ? x + 1.f : __expf(x);
}

// Packed-fragment B layout: element (k, c) of a 32x16 tile lives at
//   lane = (c & 15) | (k & 16),  slot = k & 15   (contiguous k order, same as frag_a_row)
__device__ __forceinline__ int bpack_off(int k, int c) {
  return ((c & 15) | (k & 16)) * 16 + (k & 15);
}

// ---------------- WMMA fragment loaders (CDNA5 wave32 layouts) ----------------
// A fragment 16x32 from row-major src[row*ld + k]  (contiguous -> b128 loads)
__device__ __forceinline__ v16bf frag_a_row(const bf16_t* src, int ld) {
  const int l = lane_id();
  const int row = l & 15;
  const int kb = (l >> 4) << 4;  // 0 or 16
  v16bf f;
#pragma unroll
  for (int j = 0; j < 8; ++j) f[j] = src[row * ld + kb + j];
#pragma unroll
  for (int j = 0; j < 8; ++j) f[8 + j] = src[row * ld + kb + 8 + j];
  return f;
}
// B fragment from packed-fragment storage: 32 bytes contiguous per lane
__device__ __forceinline__ v16bf frag_packed(const bf16_t* base) {
  return *(const v16bf*)(base + lane_id() * 16);
}
// B fragment 32x16 where B[k][n] = src[n*ld + k] (k^T; contiguous per lane)
__device__ __forceinline__ v16bf frag_b_trans(const bf16_t* src, int ld) {
  const int l = lane_id();
  const int col = l & 15;
  const int kb = (l >> 4) << 4;
  v16bf f;
#pragma unroll
  for (int j = 0; j < 16; ++j) f[j] = src[col * ld + kb + j];
  return f;
}
// C/D fragment 16x16 f32 store
__device__ __forceinline__ void frag_store(float* dst, int ld, v8f c) {
  const int l = lane_id();
  const int col = l & 15;
  const int rb = (l >> 4) * 8;
#pragma unroll
  for (int r = 0; r < 8; ++r) dst[(rb + r) * ld + col] = c[r];
}

// ---------------- one-time prep: X -> bf16, W -> packed bf16 ----------------
__global__ __launch_bounds__(256) void cvt_x_kernel(const float* __restrict__ X,
                                                    bf16_t* __restrict__ Xb) {
  size_t idx = ((size_t)blockIdx.x * 256 + threadIdx.x) * 4;
  float4 v = *(const float4*)(X + idx);
  Xb[idx + 0] = (bf16_t)v.x;
  Xb[idx + 1] = (bf16_t)v.y;
  Xb[idx + 2] = (bf16_t)v.z;
  Xb[idx + 3] = (bf16_t)v.w;
}
// Wpk layout: [kstep 0..31][ntile 0..63][512 packed]
__global__ __launch_bounds__(256) void pack_w_kernel(const float* __restrict__ W,
                                                     bf16_t* __restrict__ Wpk) {
  int idx = blockIdx.x * 256 + threadIdx.x;     // over DMODEL*DMODEL
  int k = idx >> 10, n = idx & 1023;
  Wpk[((size_t)(k >> 5) * 64 + (n >> 4)) * 512 + bpack_off(k, n)] = (bf16_t)W[idx];
}

// ---------------- kernel 1: Y = bf16(Xb @ W + b) ----------------
// PACKV=false: row-major output [M, DMODEL] (for Q, K)
// PACKV=true : packed-fragment output per (b,h,seg): [kstep][ntile][512] (for V)
template <bool PACKV>
__global__ __launch_bounds__(256) void proj_kernel(const bf16_t* __restrict__ Xb,
                                                   const bf16_t* __restrict__ Wpk,
                                                   const float* __restrict__ bias,
                                                   bf16_t* __restrict__ Y) {
  constexpr int TM = 128, TN = 64, TK = 32, KSTEPS = DMODEL / TK;
  __shared__ alignas(32) bf16_t Bs[2][4 * 512];   // double-buffered packed B tiles
  const int n0 = blockIdx.x * TN;
  const int m0 = blockIdx.y * TM;
  const int t = threadIdx.x;
  const int wave = t >> 5;
  const bf16_t* arow = Xb + (size_t)(m0 + wave * 16) * DMODEL;
  const bf16_t* wbase = Wpk + (size_t)(n0 >> 4) * 512;   // + ks*64*512 per step
  v8f acc[4] = {};
  if (t < 128)                                    // stage k-step 0 B tile
    *(v16bf*)(&Bs[0][t * 16]) = *(const v16bf*)(wbase + t * 16);
  __syncthreads();
  for (int ks = 0; ks < KSTEPS; ++ks) {
    if (ks + 1 < KSTEPS) {
      if (t < 128)                                // stage next B tile
        *(v16bf*)(&Bs[(ks + 1) & 1][t * 16]) =
            *(const v16bf*)(wbase + (size_t)(ks + 1) * 64 * 512 + t * 16);
      __builtin_prefetch(arow + (ks + 1) * TK + (lane_id() & 15) * DMODEL, 0, 1);
    }
    v16bf af = frag_a_row(arow + ks * TK, DMODEL);  // direct from global bf16 X
#pragma unroll
    for (int nt = 0; nt < 4; ++nt)
      acc[nt] = wmma_bf16(af, frag_packed(&Bs[ks & 1][nt * 512]), acc[nt]);
    __syncthreads();
  }
  const int lane = lane_id();
  const int col = lane & 15;
  const int rb = (lane >> 4) * 8;
#pragma unroll
  for (int nt = 0; nt < 4; ++nt) {
    int nc = n0 + nt * 16 + col;
    float bv = bias[nc];
#pragma unroll
    for (int r = 0; r < 8; ++r) {
      int m = m0 + wave * 16 + rb + r;
      bf16_t val = (bf16_t)(acc[nt][r] + bv);
      if (!PACKV) {
        Y[(size_t)m * DMODEL + nc] = val;
      } else {
        int b = m >> 13, rem = m & (SS - 1);
        int s = rem >> 9, lrow = rem & (SEGL - 1);
        int h = nc >> 7, c = nc & (DK - 1);
        size_t base = ((size_t)(b * NH + h) * NSEG + s) * (SEGL * DV);
        Y[base + ((size_t)(lrow >> 5) * 8 + (c >> 4)) * 512 + bpack_off(lrow, c)] = val;
      }
    }
  }
}

// ------- kernel 2: per-segment partial state  P = (elu(k)+1)^T @ v,  z = sums -------
__global__ __launch_bounds__(256) void kv_state_kernel(const bf16_t* __restrict__ Kb,
                                                       const bf16_t* __restrict__ Vpk,
                                                       float* __restrict__ kvpart,
                                                       float* __restrict__ zpart) {
  constexpr int LTT = 40;                       // 32 + 8, [128 c][32 l] transposed
  __shared__ bf16_t skt[DK * LTT];
  __shared__ float zred[256];
  const int bid = blockIdx.x;                   // b*NH*NSEG + h*NSEG + s
  const size_t kbase = ((size_t)(bid >> 7) * SS + (size_t)(bid & (NSEG - 1)) * SEGL) * DMODEL +
                       (size_t)((bid >> 4) & (NH - 1)) * DK;
  const bf16_t* kp = Kb + kbase;
  const bf16_t* vseg = Vpk + (size_t)bid * (SEGL * DV);
  const int t = threadIdx.x, wave = t >> 5;
  v8f acc[8] = {};
  float zacc = 0.f;
  for (int l0 = 0; l0 < SEGL; l0 += 32) {
#pragma unroll
    for (int i = 0; i < 16; ++i) {              // stage sk^T tile (transposed)
      int idx = i * 256 + t;
      int r = idx >> 7, c = idx & 127;          // r = l local, c = dk
      float sk = elup1((float)kp[(size_t)(l0 + r) * DMODEL + c]);
      zacc += sk;
      skt[c * LTT + r] = (bf16_t)sk;
    }
    __syncthreads();
#pragma unroll
    for (int nt = 0; nt < 8; ++nt)
      acc[nt] = wmma_bf16(frag_a_row(&skt[(wave * 16) * LTT], LTT),
                          frag_packed(vseg + ((l0 >> 5) * 8 + nt) * 512), acc[nt]);
    __syncthreads();
  }
  float* P = kvpart + (size_t)bid * DK * DV;
#pragma unroll
  for (int nt = 0; nt < 8; ++nt)
    frag_store(P + (wave * 16) * DV + nt * 16, DV, acc[nt]);
  zred[t] = zacc;                               // thread t always handled column t&127
  __syncthreads();
  if (t < DK)
    zpart[(size_t)bid * DK + t] = (zred[t] + zred[t + 128]) * (float)SEGL;
}

// ------- kernel 3: inclusive prefix over segments -> packed bf16 memory state -------
__global__ __launch_bounds__(256) void prefix_kv_kernel(const float* __restrict__ kv,
                                                        bf16_t* __restrict__ mp) {
  size_t idx = (size_t)blockIdx.x * 256 + threadIdx.x;   // BB*NH*DK*DV chains
  size_t bh = idx >> 14;
  int e = (int)(idx & 16383);
  int ki = e >> 7, j = e & 127;                 // dk row, dv col
  int poff = ((ki >> 5) * 8 + (j >> 4)) * 512 + bpack_off(ki, j);
  float run = 0.f;
#pragma unroll
  for (int s = 0; s < NSEG; ++s) {
    run += kv[(bh * NSEG + s) * (size_t)(DK * DV) + e];
    mp[(bh * NSEG + s) * (size_t)(DK * DV) + poff] = (bf16_t)run;
  }
}
__global__ __launch_bounds__(256) void prefix_z_kernel(float* __restrict__ z) {
  int idx = blockIdx.x * 256 + threadIdx.x;
  if (idx >= BB * NH * DK) return;
  int bh = idx >> 7;
  int c = idx & 127;
  float run = 0.f;
#pragma unroll
  for (int s = 0; s < NSEG; ++s) {
    size_t off = ((size_t)bh * NSEG + s) * DK + c;
    run += z[off];
    z[off] = run;
  }
}

// ------- kernel 4: retrieval + local softmax attention + head-mean blend -------
__global__ __launch_bounds__(256) void attn_kernel(const bf16_t* __restrict__ Qb,
                                                   const bf16_t* __restrict__ Kb,
                                                   const bf16_t* __restrict__ Vpk,
                                                   const bf16_t* __restrict__ mempk,
                                                   const float* __restrict__ zstate,
                                                   const float* __restrict__ beta,
                                                   float* __restrict__ out) {
  constexpr int LQ = DK + 8, LSC = SEGL + 8, LPS = SEGL + 16;
  __shared__ bf16_t qs[32 * LQ];
  __shared__ bf16_t sqs[32 * LQ];
  __shared__ float sc[32 * LSC];      // f32 scores 32x512
  __shared__ bf16_t ps[32 * LPS];     // normalized probs bf16
  __shared__ float denom[32];
  __shared__ float red[32 * 8];
  __shared__ float rowstat[32];
  const int bid = blockIdx.x;         // b*256 + s*16 + lt
  const int lt = bid & 15;
  const int s = (bid >> 4) & 15;
  const int b = bid >> 8;
  const int t = threadIdx.x, wave = t >> 5, lane = t & 31;
  const int mstrip = wave >> 2;       // output row strip (0..1)
  const int npair = wave & 3;         // output col tile pair
  const int srow = (wave >> 2) * 16;  // score row base
  const int snc0 = (wave & 3) * 128;  // score col range
  const float bsig = 1.f / (1.f + __expf(-beta[0]));
  const float cmem = bsig * 0.125f;
  const float cdot = (1.f - bsig) * 0.125f;
  const float scale = 0.088388347648318447f;  // 1/sqrt(128)
  const size_t tok0 = (size_t)b * SS + (size_t)s * SEGL + (size_t)lt * 32;
  const size_t segbase = ((size_t)b * SS + (size_t)s * SEGL) * DMODEL;
  v8f oacc[2] = {};
  for (int h = 0; h < NH; ++h) {
    const bf16_t* qp = Qb + tok0 * DMODEL + (size_t)h * DK;
    const bf16_t* kp = Kb + segbase + (size_t)h * DK;
    const size_t bhs = (size_t)(b * NH + h) * NSEG + s;
    const bf16_t* vp = Vpk + bhs * (SEGL * DV);
    const bf16_t* memb = mempk + bhs * (DK * DV);
    const float* zb = zstate + bhs * DK;
#pragma unroll
    for (int i = 0; i < 16; ++i) {    // stage q and sq = elu(q)+1
      int idx = i * 256 + t;
      int r = idx >> 7, c = idx & 127;
      bf16_t qv = qp[(size_t)r * DMODEL + c];
      qs[r * LQ + c] = qv;
      sqs[r * LQ + c] = (bf16_t)elup1((float)qv);
    }
    __syncthreads();
    if (t < 32) {                     // per-row denominator sq . z + 1e-5
      float d = 1e-5f;
      for (int c = 0; c < DK; ++c) d += (float)sqs[t * LQ + c] * zb[c];
      denom[t] = d;
    }
    // retrieval numerator: sq(32x128) @ mem(128x128), packed-fragment B
    v8f mm[2] = {};
#pragma unroll
    for (int j = 0; j < 2; ++j) {
      int nt = npair * 2 + j;
#pragma unroll
      for (int kk = 0; kk < 4; ++kk)
        mm[j] = wmma_bf16(frag_a_row(&sqs[(mstrip * 16) * LQ + kk * 32], LQ),
                          frag_packed(memb + (kk * 8 + nt) * 512), mm[j]);
    }
    // scores: q(32x128) @ k^T(128x512)
    for (int nt = 0; nt < 8; ++nt) {
      v8f s8 = {};
#pragma unroll
      for (int kk = 0; kk < DK; kk += 32)
        s8 = wmma_bf16(frag_a_row(&qs[srow * LQ + kk], LQ),
                       frag_b_trans(kp + (size_t)(snc0 + nt * 16) * DMODEL + kk,
                                    DMODEL), s8);
      frag_store(&sc[srow * LSC + snc0 + nt * 16], LSC, s8);
    }
    __syncthreads();
    // row softmax, 8 threads per row
    {
      int r8 = t >> 3, su = t & 7;
      float lm = -1e30f;
      for (int i = 0; i < 64; ++i)
        lm = fmaxf(lm, sc[r8 * LSC + su * 64 + i] * scale);
      red[r8 * 8 + su] = lm;
      __syncthreads();
      if (su == 0) {
        float m2 = red[r8 * 8];
        for (int j2 = 1; j2 < 8; ++j2) m2 = fmaxf(m2, red[r8 * 8 + j2]);
        rowstat[r8] = m2;
      }
      __syncthreads();
      float mx = rowstat[r8], ls = 0.f;
      for (int i = 0; i < 64; ++i) {
        float e = __expf(sc[r8 * LSC + su * 64 + i] * scale - mx);
        sc[r8 * LSC + su * 64 + i] = e;
        ls += e;
      }
      red[r8 * 8 + su] = ls;
      __syncthreads();
      if (su == 0) {
        float sm = red[r8 * 8];
        for (int j2 = 1; j2 < 8; ++j2) sm += red[r8 * 8 + j2];
        rowstat[r8] = 1.f / sm;
      }
      __syncthreads();
      float inv = rowstat[r8];
      for (int i = 0; i < 64; ++i)
        ps[r8 * LPS + su * 64 + i] = (bf16_t)(sc[r8 * LSC + su * 64 + i] * inv);
    }
    __syncthreads();
    // a_dot: probs(32x512) @ v(512x128), packed-fragment B
    v8f dd[2] = {};
#pragma unroll
    for (int j = 0; j < 2; ++j) {
      int nt = npair * 2 + j;
#pragma unroll
      for (int kk = 0; kk < 16; ++kk)
        dd[j] = wmma_bf16(frag_a_row(&ps[(mstrip * 16) * LPS + kk * 32], LPS),
                          frag_packed(vp + (kk * 8 + nt) * 512), dd[j]);
    }
    // blend into persistent output fragments (head mean folded into coeffs)
    {
      int rb = (lane >> 4) * 8;
#pragma unroll
      for (int j = 0; j < 2; ++j)
#pragma unroll
        for (int r = 0; r < 8; ++r) {
          int row = mstrip * 16 + rb + r;
          oacc[j][r] += cmem * (mm[j][r] / denom[row]) + cdot * dd[j][r];
        }
    }
    __syncthreads();
  }
  float* ob = out + tok0 * DV;
#pragma unroll
  for (int j = 0; j < 2; ++j)
    frag_store(ob + (size_t)(mstrip * 16) * DV + (npair * 2 + j) * 16, DV, oacc[j]);
}

// ---------------- host launcher ----------------
extern "C" void kernel_launch(void* const* d_in, const int* in_sizes, int n_in,
                              void* d_out, int out_size, void* d_ws, size_t ws_size,
                              hipStream_t stream) {
  (void)in_sizes; (void)n_in; (void)out_size; (void)ws_size;
  const float* x = (const float*)d_in[0];
  const float* wq = (const float*)d_in[1];
  const float* bq = (const float*)d_in[2];
  const float* wk = (const float*)d_in[3];
  const float* bk = (const float*)d_in[4];
  const float* wv = (const float*)d_in[5];
  const float* bv = (const float*)d_in[6];
  const float* beta = (const float*)d_in[7];
  float* out = (float*)d_out;

  char* ws = (char*)d_ws;
  const size_t qkv_bytes = (size_t)MTOT * DMODEL * sizeof(bf16_t);   // 64 MB each
  const size_t kv_bytes = (size_t)BB * NH * NSEG * DK * DV * sizeof(float);  // 32 MB
  const size_t w_bytes = (size_t)DMODEL * DMODEL * sizeof(bf16_t);   // 2 MB each
  bf16_t* Qb = (bf16_t*)ws;
  bf16_t* Kb = (bf16_t*)(ws + qkv_bytes);
  bf16_t* Vpk = (bf16_t*)(ws + 2 * qkv_bytes);           // packed-fragment V
  float* kvpart = (float*)(ws + 3 * qkv_bytes);
  bf16_t* mempk = (bf16_t*)(ws + 3 * qkv_bytes + kv_bytes);
  float* zstate = (float*)(ws + 3 * qkv_bytes + kv_bytes + kv_bytes / 2);
  char* p2 = ws + 3 * qkv_bytes + 2 * kv_bytes;          // (z uses far less; safe)
  bf16_t* Xb = (bf16_t*)p2;                              // bf16 copy of x (64 MB)
  bf16_t* Wq = (bf16_t*)(p2 + qkv_bytes);
  bf16_t* Wk = (bf16_t*)(p2 + qkv_bytes + w_bytes);
  bf16_t* Wv = (bf16_t*)(p2 + qkv_bytes + 2 * w_bytes);

  cvt_x_kernel<<<(MTOT * (DMODEL / 4)) / 256, 256, 0, stream>>>(x, Xb);
  pack_w_kernel<<<(DMODEL * DMODEL) / 256, 256, 0, stream>>>(wq, Wq);
  pack_w_kernel<<<(DMODEL * DMODEL) / 256, 256, 0, stream>>>(wk, Wk);
  pack_w_kernel<<<(DMODEL * DMODEL) / 256, 256, 0, stream>>>(wv, Wv);

  dim3 pg(DMODEL / 64, MTOT / 128);
  proj_kernel<false><<<pg, 256, 0, stream>>>(Xb, Wq, bq, Qb);
  proj_kernel<false><<<pg, 256, 0, stream>>>(Xb, Wk, bk, Kb);
  proj_kernel<true><<<pg, 256, 0, stream>>>(Xb, Wv, bv, Vpk);
  kv_state_kernel<<<BB * NH * NSEG, 256, 0, stream>>>(Kb, Vpk, kvpart, zstate);
  prefix_kv_kernel<<<(BB * NH * DK * DV) / 256, 256, 0, stream>>>(kvpart, mempk);
  prefix_z_kernel<<<(BB * NH * DK + 255) / 256, 256, 0, stream>>>(zstate);
  attn_kernel<<<BB * NSEG * 16, 256, 0, stream>>>(Qb, Kb, Vpk, mempk, zstate,
                                                  beta, out);
}
